// SparseAutoencoder_78494822302182
// MI455X (gfx1250) — matrix operand order repaired and checked by reading, not compiled, not run on hardware
//
#include <hip/hip_runtime.h>
#include <hip/hip_bf16.h>

#define BB 4096
#define DD 1024
#define MM 16384
#define K_TOT 131072u   // K_ACTIVE * B
#define TOPR 64         // per-row top-k (MULTI_K == AUX_K == 64)

typedef __attribute__((ext_vector_type(16))) __bf16 v16bf;
typedef __attribute__((ext_vector_type(8)))  __bf16 v8bf;
typedef __attribute__((ext_vector_type(8)))  float  v8f;
typedef int v4si __attribute__((vector_size(16)));

// ------------------------------------------------ CDNA5 async-to-LDS support
#if defined(__has_builtin)
#if __has_builtin(__builtin_amdgcn_global_load_async_to_lds_b128)
#define GEMM_ASYNC 1
#endif
#endif

#ifdef GEMM_ASYNC
// Builtin signature: (v4si addrspace(1)* src, v4si addrspace(3)* dst, Imm offset, Imm cpol)
// INST_OFFSET is added to BOTH the global and LDS address (ISA 08 §4.4)
#define ASYNC16(g, l, ofs)                                                     \
    __builtin_amdgcn_global_load_async_to_lds_b128(                            \
        (__attribute__((address_space(1))) v4si*)(g),                          \
        (__attribute__((address_space(3))) v4si*)(l), (ofs), 0)

__device__ __forceinline__ void wait_async0() {
#if __has_builtin(__builtin_amdgcn_s_wait_asynccnt)
    __builtin_amdgcn_s_wait_asynccnt(0);
#else
    asm volatile("s_wait_asynccnt 0x0" ::: "memory");
#endif
}
#endif

// ---------------------------------------------------------------- helpers
__device__ __forceinline__ unsigned fkey(float f) {
    unsigned u = __float_as_uint(f);
    return (u & 0x80000000u) ? ~u : (u | 0x80000000u);  // total order, asc
}

__device__ __forceinline__ unsigned exscan256(unsigned v, unsigned* buf, int tid) {
    unsigned x = v;
    buf[tid] = x; __syncthreads();
    for (int o = 1; o < 256; o <<= 1) {
        unsigned t = (tid >= o) ? buf[tid - o] : 0u;
        __syncthreads();
        x += t; buf[tid] = x;
        __syncthreads();
    }
    return x - v;  // exclusive prefix
}

// ---------------------------------------------------------------- prep
__global__ __launch_bounds__(256) void k_prep_x(const float* __restrict__ x,
                                                const float* __restrict__ ibias,
                                                __bf16* __restrict__ xb) {
    int i = blockIdx.x * 256 + threadIdx.x;      // B*D total
    int d = i & (DD - 1);
    xb[i] = (__bf16)(x[i] - ibias[d]);
}

__global__ __launch_bounds__(256) void k_conv_w(const float* __restrict__ w,
                                                __bf16* __restrict__ wb) {
    int i = blockIdx.x * 256 + threadIdx.x;      // M*D total
    wb[i] = (__bf16)w[i];
}

__global__ __launch_bounds__(256) void k_transpose_dec(const float* __restrict__ src, // [D][M]
                                                       float* __restrict__ dst) {     // [M][D]
    __shared__ float tile[32][33];
    int mBlk = blockIdx.x * 32, dBlk = blockIdx.y * 32;
    int tx = threadIdx.x, ty = threadIdx.y;      // block (32,8)
    for (int i = ty; i < 32; i += 8)
        tile[i][tx] = src[(size_t)(dBlk + i) * MM + mBlk + tx];
    __syncthreads();
    for (int i = ty; i < 32; i += 8)
        dst[(size_t)(mBlk + i) * DD + dBlk + tx] = tile[tx][i];
}

// ---------------------------------------------------------------- encoder GEMM (WMMA bf16)
// pre[B][M] = xb[B][D] @ wb[M][D]^T + neuron_bias
// 128x128 block tile, 8 waves, each wave 64x32 (4x2 accumulators),
// double-buffered LDS staged with GLOBAL_LOAD_ASYNC_TO_LDS_B128.
__global__ __launch_bounds__(256) void k_gemm_enc(const __bf16* __restrict__ xb,
                                                  const __bf16* __restrict__ wb,
                                                  const float* __restrict__ nbias,
                                                  float* __restrict__ pre) {
    __shared__ __bf16 As[2][128 * 32];   // [buf][row][k]
    __shared__ __bf16 Bs[2][128 * 32];   // [buf][col][k]
    const int rowBlk = blockIdx.y * 128;   // over B
    const int colBlk = blockIdx.x * 128;   // over M
    const int tid  = threadIdx.x;
    const int lane = tid & 31, wave = tid >> 5;
    const int wr = wave & 1;               // 2 row-waves  (64 rows each)
    const int wc = wave >> 1;              // 4 col-waves  (32 cols each)

    v8f acc[4][2];
    for (int i = 0; i < 4; i++)
        for (int j = 0; j < 2; j++)
            for (int e = 0; e < 8; e++) acc[i][j][e] = 0.0f;

    // staging: thread t owns 16 bf16 (32B) for A and B: row = t/2, half = t%2
    const int ldr = tid >> 1;
    const int ldc = (tid & 1) * 16;
    const __bf16* gA = xb + (size_t)(rowBlk + ldr) * DD + ldc;
    const __bf16* gB = wb + (size_t)(colBlk + ldr) * DD + ldc;

#ifdef GEMM_ASYNC
    auto issueTile = [&](int k0, int buf) {
        ASYNC16(gA + k0, &As[buf][ldr * 32 + ldc], 0);
        ASYNC16(gA + k0, &As[buf][ldr * 32 + ldc], 16);
        ASYNC16(gB + k0, &Bs[buf][ldr * 32 + ldc], 0);
        ASYNC16(gB + k0, &Bs[buf][ldr * 32 + ldc], 16);
    };
    issueTile(0, 0);
#endif

    int buf = 0;
    for (int k0 = 0; k0 < DD; k0 += 32, buf ^= 1) {
#ifdef GEMM_ASYNC
        wait_async0();            // tile k resident in LDS (this wave's chunks)
        __syncthreads();          // all waves' chunks resident
        if (k0 + 32 < DD) issueTile(k0 + 32, buf ^ 1);   // prefetch next tile
        const int cbuf = buf;
#else
        uint4 a0 = *(const uint4*)(gA + k0);
        uint4 a1 = *(const uint4*)(gA + k0 + 8);
        uint4 b0 = *(const uint4*)(gB + k0);
        uint4 b1 = *(const uint4*)(gB + k0 + 8);
        __builtin_prefetch(gA + k0 + 128, 0, 0);
        __builtin_prefetch(gB + k0 + 128, 0, 0);
        __syncthreads();
        *(uint4*)(&As[0][ldr * 32 + ldc])     = a0;
        *(uint4*)(&As[0][ldr * 32 + ldc + 8]) = a1;
        *(uint4*)(&Bs[0][ldr * 32 + ldc])     = b0;
        *(uint4*)(&Bs[0][ldr * 32 + ldc + 8]) = b1;
        __syncthreads();
        const int cbuf = 0;
#endif
        // Load ALL fragments first so the 12 ds_load_b128 issue back-to-back
        // and the WMMAs drain them with staggered s_wait_dscnt (no serialized
        // full-zero waits from register reuse).
        v16bf bfrag[2];
        {
            const int kb = (lane >> 4) * 16;
            for (int tj = 0; tj < 2; tj++) {
                int col = wc * 32 + tj * 16 + (lane & 15);
                bfrag[tj] = *(const v16bf*)(&Bs[cbuf][col * 32 + kb]);
            }
        }
        v16bf afrag[4];
        {
            const int ab = (lane >> 4) * 16;
            for (int ti = 0; ti < 4; ti++) {
                int row = wr * 64 + ti * 16 + (lane & 15);
                v8bf lo = *(const v8bf*)(&As[cbuf][row * 32 + ab]);
                v8bf hi = *(const v8bf*)(&As[cbuf][row * 32 + ab + 8]);
                afrag[ti] = __builtin_shufflevector(lo, hi,
                    0, 1, 2, 3, 4, 5, 6, 7, 8, 9, 10, 11, 12, 13, 14, 15);
            }
        }
        for (int ti = 0; ti < 4; ti++)
            for (int tj = 0; tj < 2; tj++)
                acc[ti][tj] = __builtin_amdgcn_wmma_f32_16x16x32_bf16(
                    false, afrag[ti], false, bfrag[tj], (short)0, acc[ti][tj], false, false);
    }

    // epilogue: C layout — elem i: row = base + i + 8*(lane>=16), col = lane%16
    for (int ti = 0; ti < 4; ti++) {
        for (int tj = 0; tj < 2; tj++) {
            int col = colBlk + wc * 32 + tj * 16 + (lane & 15);
            float nb = nbias[col];
            int rbase = rowBlk + wr * 64 + ti * 16 + ((lane >> 4) * 8);
            for (int i = 0; i < 8; i++)
                pre[(size_t)(rbase + i) * MM + col] = acc[ti][tj][i] + nb;
        }
    }
}

// ---------------------------------------------------------------- global top-K radix select
__global__ __launch_bounds__(256) void k_hist1(const float* __restrict__ pre,
                                               unsigned* __restrict__ gh) {
    __shared__ unsigned h[4096];
    int tid = threadIdx.x;
    for (int i = tid; i < 4096; i += 256) h[i] = 0u;
    __syncthreads();
    size_t N4 = (size_t)BB * MM / 4;
    const float4* p4 = (const float4*)pre;
    for (size_t i = (size_t)blockIdx.x * 256 + tid; i < N4; i += (size_t)gridDim.x * 256) {
        float4 v = p4[i];
        if (v.x > 0.0f) atomicAdd(&h[__float_as_uint(v.x) >> 19], 1u);
        if (v.y > 0.0f) atomicAdd(&h[__float_as_uint(v.y) >> 19], 1u);
        if (v.z > 0.0f) atomicAdd(&h[__float_as_uint(v.z) >> 19], 1u);
        if (v.w > 0.0f) atomicAdd(&h[__float_as_uint(v.w) >> 19], 1u);
    }
    __syncthreads();
    for (int i = tid; i < 4096; i += 256)
        if (h[i]) atomicAdd(&gh[i], h[i]);
}

__global__ void k_select1(const unsigned* __restrict__ gh, unsigned* __restrict__ sel) {
    if (threadIdx.x == 0 && blockIdx.x == 0) {
        unsigned cum = 0; int b = 4095;
        for (; b > 0; b--) { unsigned c = gh[b]; if (cum + c >= K_TOT) break; cum += c; }
        sel[0] = (unsigned)b; sel[1] = cum;
    }
}

__global__ __launch_bounds__(256) void k_hist2(const float* __restrict__ pre,
                                               const unsigned* __restrict__ sel,
                                               unsigned* __restrict__ gh2) {
    __shared__ unsigned h[4096];
    int tid = threadIdx.x;
    for (int i = tid; i < 4096; i += 256) h[i] = 0u;
    __syncthreads();
    unsigned b1 = sel[0];
    size_t N4 = (size_t)BB * MM / 4;
    const float4* p4 = (const float4*)pre;
    for (size_t i = (size_t)blockIdx.x * 256 + tid; i < N4; i += (size_t)gridDim.x * 256) {
        float4 v = p4[i];
        unsigned u;
        u = __float_as_uint(v.x); if (v.x > 0.0f && (u >> 19) == b1) atomicAdd(&h[(u >> 7) & 0xFFFu], 1u);
        u = __float_as_uint(v.y); if (v.y > 0.0f && (u >> 19) == b1) atomicAdd(&h[(u >> 7) & 0xFFFu], 1u);
        u = __float_as_uint(v.z); if (v.z > 0.0f && (u >> 19) == b1) atomicAdd(&h[(u >> 7) & 0xFFFu], 1u);
        u = __float_as_uint(v.w); if (v.w > 0.0f && (u >> 19) == b1) atomicAdd(&h[(u >> 7) & 0xFFFu], 1u);
    }
    __syncthreads();
    for (int i = tid; i < 4096; i += 256)
        if (h[i]) atomicAdd(&gh2[i], h[i]);
}

__global__ void k_select2(const unsigned* __restrict__ gh2, unsigned* __restrict__ sel) {
    if (threadIdx.x == 0 && blockIdx.x == 0) {
        unsigned cum = sel[1]; int b = 4095;
        for (; b > 0; b--) { unsigned c = gh2[b]; if (cum + c >= K_TOT) break; cum += c; }
        sel[2] = (sel[0] << 19) | ((unsigned)b << 7);   // threshold bit pattern
    }
}

// ------------------------------------- fused: threshold scatter + active flags + sparse recon
__global__ __launch_bounds__(256) void k_row_scatter_recon(
        const float* __restrict__ pre, const unsigned* __restrict__ sel,
        const float* __restrict__ dec_wT, const float* __restrict__ ibias,
        float* __restrict__ activ, unsigned* __restrict__ active,
        float* __restrict__ recon) {
    __shared__ float cval[1024];
    __shared__ int   cidx[1024];
    __shared__ unsigned sbuf[256];
    __shared__ unsigned totalS;
    const int r = blockIdx.x, tid = threadIdx.x;
    const float* row = pre + (size_t)r * MM;
    float* arow = activ + (size_t)r * MM;
    const unsigned th = sel[2];
    const int base = tid * 64;

    unsigned n = 0;
    for (int i = 0; i < 64; i += 4) {
        float4 v = *(const float4*)(row + base + i);
        float4 o;
        o.x = (v.x > 0.0f && __float_as_uint(v.x) >= th) ? v.x : 0.0f;
        o.y = (v.y > 0.0f && __float_as_uint(v.y) >= th) ? v.y : 0.0f;
        o.z = (v.z > 0.0f && __float_as_uint(v.z) >= th) ? v.z : 0.0f;
        o.w = (v.w > 0.0f && __float_as_uint(v.w) >= th) ? v.w : 0.0f;
        *(float4*)(arow + base + i) = o;
        if (o.x != 0.0f) { n++; active[base + i + 0] = 1u; }
        if (o.y != 0.0f) { n++; active[base + i + 1] = 1u; }
        if (o.z != 0.0f) { n++; active[base + i + 2] = 1u; }
        if (o.w != 0.0f) { n++; active[base + i + 3] = 1u; }
    }
    unsigned off = exscan256(n, sbuf, tid);
    if (tid == 255) totalS = off + n;
    __syncthreads();
    unsigned total = totalS; if (total > 1024u) total = 1024u;
    unsigned o = off;
    for (int i = 0; i < 64; i++) {
        float v = row[base + i];
        float kept = (v > 0.0f && __float_as_uint(v) >= th) ? v : 0.0f;
        if (kept != 0.0f) { if (o < 1024u) { cval[o] = kept; cidx[o] = base + i; } o++; }
    }
    __syncthreads();

    int d0 = tid * 4;
    float4 acc = *(const float4*)(ibias + d0);
    for (unsigned j = 0; j < total; j++) {
        float v = cval[j];
        float4 w = *(const float4*)(dec_wT + (size_t)cidx[j] * DD + d0);
        acc.x += v * w.x; acc.y += v * w.y; acc.z += v * w.z; acc.w += v * w.w;
    }
    *(float4*)(recon + (size_t)r * DD + d0) = acc;
}

__global__ __launch_bounds__(256) void k_steps_dead(const int* __restrict__ steps,
                                                    const unsigned* __restrict__ active,
                                                    float* __restrict__ dead) {
    int m = blockIdx.x * 256 + threadIdx.x;
    int s = steps[m] + 1;
    if (active[m]) s = 0;
    dead[m] = (s > 256) ? 1.0f : 0.0f;
}

// ------------------------------------- fused per-row top-64: multik decode + aux outputs
__global__ __launch_bounds__(256) void k_row_topk2(const float* __restrict__ pre,
                                                   const float* __restrict__ dead,
                                                   const float* __restrict__ dec_wT,
                                                   const float* __restrict__ ibias,
                                                   float* __restrict__ mk_recon,
                                                   float* __restrict__ aux_vals,
                                                   int* __restrict__ aux_idx) {
    __shared__ unsigned hist[256];
    __shared__ unsigned sbuf[256];
    __shared__ unsigned sP[4];
    __shared__ float sval[TOPR];
    __shared__ int   sidx[TOPR];

    const int r = blockIdx.x;
    const int tid = threadIdx.x;
    const float* row = pre + (size_t)r * MM;
    const int base = tid * 64;                  // blocked ownership -> deterministic

    for (int mode = 0; mode < 2; ++mode) {
        // pass 1: hist on top byte of order-key
        hist[tid] = 0u; __syncthreads();
        for (int i = 0; i < 64; i++) {
            float v = row[base + i]; if (mode) v *= dead[base + i];
            atomicAdd(&hist[fkey(v) >> 24], 1u);
        }
        __syncthreads();
        if (tid == 0) {
            unsigned cum = 0; int b = 255;
            for (; b > 0; b--) { unsigned c = hist[b]; if (cum + c >= TOPR) break; cum += c; }
            sP[0] = (unsigned)b; sP[1] = cum;
        }
        __syncthreads();
        const unsigned b1 = sP[0], cnt1 = sP[1];

        // pass 2: hist on second byte among top-byte == b1
        hist[tid] = 0u; __syncthreads();
        for (int i = 0; i < 64; i++) {
            float v = row[base + i]; if (mode) v *= dead[base + i];
            unsigned k = fkey(v);
            if ((k >> 24) == b1) atomicAdd(&hist[(k >> 16) & 255u], 1u);
        }
        __syncthreads();
        if (tid == 0) {
            unsigned cum = cnt1; int b = 255;
            for (; b > 0; b--) { unsigned c = hist[b]; if (cum + c >= TOPR) break; cum += c; }
            sP[2] = (b1 << 8) | (unsigned)b; sP[3] = cum;
        }
        __syncthreads();
        const unsigned t16 = sP[2], cntAbove = sP[3];   // cntAbove < 64

        // deterministic selection: greater elements first, then index-ordered ties
        unsigned ng = 0, nt = 0;
        for (int i = 0; i < 64; i++) {
            float v = row[base + i]; if (mode) v *= dead[base + i];
            unsigned hi = fkey(v) >> 16;
            if (hi > t16) ng++; else if (hi == t16) nt++;
        }
        unsigned gOff = exscan256(ng, sbuf, tid);
        unsigned tOff = exscan256(nt, sbuf, tid);
        unsigned g = gOff, t = cntAbove + tOff;
        for (int i = 0; i < 64; i++) {
            float v = row[base + i]; if (mode) v *= dead[base + i];
            unsigned hi = fkey(v) >> 16;
            if (hi > t16)       { sval[g] = v; sidx[g] = base + i; g++; }
            else if (hi == t16) { if (t < TOPR) { sval[t] = v; sidx[t] = base + i; } t++; }
        }
        __syncthreads();

        // bitonic sort 64, descending by value, ascending index tiebreak
        for (int k = 2; k <= TOPR; k <<= 1) {
            for (int j = k >> 1; j > 0; j >>= 1) {
                __syncthreads();
                if (tid < TOPR) {
                    int ixj = tid ^ j;
                    if (ixj > tid) {
                        float a = sval[tid], bb = sval[ixj];
                        int ia = sidx[tid], ib = sidx[ixj];
                        bool desc = ((tid & k) == 0);
                        bool sw = desc ? ((a < bb) || (a == bb && ia > ib))
                                       : ((a > bb) || (a == bb && ia < ib));
                        if (sw) { sval[tid] = bb; sval[ixj] = a; sidx[tid] = ib; sidx[ixj] = ia; }
                    }
                }
            }
        }
        __syncthreads();

        if (mode == 1) {
            if (tid < TOPR) {
                float v = sval[tid];
                aux_vals[(size_t)r * TOPR + tid] = v > 0.0f ? v : 0.0f;
                aux_idx [(size_t)r * TOPR + tid] = sidx[tid];
            }
        } else {
            // sparse decode: mk_recon[r] = sum_j relu(val_j) * dec_wT[idx_j][:] + ibias
            int d0 = tid * 4;
            float4 acc = *(const float4*)(ibias + d0);
            for (int j = 0; j < TOPR; j++) {
                float v = sval[j]; v = v > 0.0f ? v : 0.0f;
                if (v != 0.0f) {
                    float4 w = *(const float4*)(dec_wT + (size_t)sidx[j] * DD + d0);
                    acc.x += v * w.x; acc.y += v * w.y; acc.z += v * w.z; acc.w += v * w.w;
                }
            }
            *(float4*)(mk_recon + (size_t)r * DD + d0) = acc;
        }
        __syncthreads();
    }
}

// ---------------------------------------------------------------- host launcher
extern "C" void kernel_launch(void* const* d_in, const int* in_sizes, int n_in,
                              void* d_out, int out_size, void* d_ws, size_t ws_size,
                              hipStream_t stream) {
    const float* x      = (const float*)d_in[0];
    const float* enc_w  = (const float*)d_in[1];
    const float* dec_w  = (const float*)d_in[2];
    const float* ibias  = (const float*)d_in[3];
    const float* nbias  = (const float*)d_in[4];
    const int*   steps  = (const int*)d_in[5];

    // workspace layout
    char* ws = (char*)d_ws;
    const size_t off_xb    = 0;
    const size_t off_wb    = off_xb  + (size_t)BB * DD * 2;
    const size_t off_pre   = off_wb  + (size_t)MM * DD * 2;
    const size_t off_dwt   = off_pre + (size_t)BB * MM * 4;
    const size_t off_h1    = off_dwt + (size_t)MM * DD * 4;
    const size_t off_h2    = off_h1  + 4096 * 4;
    const size_t off_sel   = off_h2  + 4096 * 4;
    const size_t off_act   = off_sel + 64;
    const size_t off_dead  = off_act + (size_t)MM * 4;

    __bf16*   xb     = (__bf16*)(ws + off_xb);
    __bf16*   wb     = (__bf16*)(ws + off_wb);
    float*    pre    = (float*)(ws + off_pre);
    float*    dwt    = (float*)(ws + off_dwt);
    unsigned* h1     = (unsigned*)(ws + off_h1);
    unsigned* h2     = (unsigned*)(ws + off_h2);
    unsigned* sel    = (unsigned*)(ws + off_sel);
    unsigned* active = (unsigned*)(ws + off_act);
    float*    dead   = (float*)(ws + off_dead);

    // outputs (concatenated in reference return order)
    float* out      = (float*)d_out;
    float* o_recon  = out;                                        // B*D
    float* o_activ  = out + (size_t)BB * DD;                      // B*M
    float* o_mk     = o_activ + (size_t)BB * MM;                  // B*D
    float* o_av     = o_mk + (size_t)BB * DD;                     // B*64
    int*   o_ai     = (int*)(o_av + (size_t)BB * TOPR);           // B*64

    // zero histograms / selection / active flags
    (void)hipMemsetAsync(ws + off_h1, 0, 4096 * 4 * 2 + 64 + (size_t)MM * 4, stream);

    k_prep_x<<<(BB * DD) / 256, 256, 0, stream>>>(x, ibias, xb);
    k_conv_w<<<(MM * DD) / 256, 256, 0, stream>>>(enc_w, wb);
    k_transpose_dec<<<dim3(MM / 32, DD / 32), dim3(32, 8), 0, stream>>>(dec_w, dwt);

    k_gemm_enc<<<dim3(MM / 128, BB / 128), 256, 0, stream>>>(xb, wb, nbias, pre);

    k_hist1<<<2048, 256, 0, stream>>>(pre, h1);
    k_select1<<<1, 64, 0, stream>>>(h1, sel);
    k_hist2<<<2048, 256, 0, stream>>>(pre, sel, h2);
    k_select2<<<1, 64, 0, stream>>>(h2, sel);

    k_row_scatter_recon<<<BB, 256, 0, stream>>>(pre, sel, dwt, ibias,
                                                o_activ, active, o_recon);

    k_steps_dead<<<MM / 256, 256, 0, stream>>>(steps, active, dead);

    k_row_topk2<<<BB, 256, 0, stream>>>(pre, dead, dwt, ibias, o_mk, o_av, o_ai);
}
